// ImMatchNet_59974923321480
// MI455X (gfx1250) — matrix operand order repaired and checked by reading, not compiled, not run on hardware
//
#include <hip/hip_runtime.h>
#include <hip/hip_bf16.h>
#include <stdint.h>

// ---------------- types ----------------
typedef _Float16 v16h __attribute__((ext_vector_type(16)));
typedef float    v8f  __attribute__((ext_vector_type(8)));
typedef _Float16 h8   __attribute__((ext_vector_type(8)));
typedef int      v4i  __attribute__((ext_vector_type(4)));

#define EPS_NORM 1e-6f
#define EPS_MM   1e-5f

#define GLOBAL_AS __attribute__((address_space(1)))
#define LDS_AS    __attribute__((address_space(3)))

#if defined(__AMDGCN__) && __has_builtin(__builtin_amdgcn_global_load_async_to_lds_b128)
#define USE_ASYNC_LDS 1
#else
#define USE_ASYNC_LDS 0
#endif

static __device__ __forceinline__ void wait_async0() {
#if __has_builtin(__builtin_amdgcn_s_wait_asynccnt)
  __builtin_amdgcn_s_wait_asynccnt(0);
#elif defined(__AMDGCN__)
  asm volatile("s_wait_asynccnt 0x0" ::: "memory");
#endif
}

static __device__ __forceinline__ v16h make_frag(h8 lo, h8 hi) {
  union { v16h v; h8 h[2]; } u;
  u.h[0] = lo; u.h[1] = hi;
  return u.v;
}

// ---------------- 1) L2 norms + f16 copies ----------------
// grid = 8 (2 tensors x 4 batches), block = 256 (one thread per spatial p)
__global__ void k_norm(const float* __restrict__ fa, const float* __restrict__ fb,
                       _Float16* __restrict__ Ah, _Float16* __restrict__ Bh,
                       float* __restrict__ invA, float* __restrict__ invB) {
  int b   = blockIdx.x & 3;
  int sel = blockIdx.x >> 2;
  const float* src = sel ? fb : fa;
  _Float16*    dst = sel ? Bh : Ah;
  float*       inv = sel ? invB : invA;
  int p = threadIdx.x;
  size_t base = (size_t)b * 1024 * 256 + p;
  float s = 0.f;
  for (int c = 0; c < 1024; ++c) {
    float v = src[base + (size_t)c * 256];
    s += v * v;
    dst[base + (size_t)c * 256] = (_Float16)v;
  }
  inv[b * 256 + p] = rsqrtf(s + EPS_NORM);
}

// ---------------- 2) correlation GEMM (WMMA), fused norm scaling ----------------
// per batch: corr[p,q] = invA[p]*invB[q] * sum_c A[c,p]*B[c,q]
// grid = B*16 blocks (16 p-row tiles), block = 512 (16 waves, one q-tile each)
__global__ void __launch_bounds__(512) k_corr(const _Float16* __restrict__ Ah,
                                              const _Float16* __restrict__ Bh,
                                              const float* __restrict__ invA,
                                              const float* __restrict__ invB,
                                              float* __restrict__ corr) {
  __shared__ __align__(16) _Float16 sA[16 * 32];   // [m][k]
  __shared__ __align__(16) _Float16 sB[256 * 32];  // [q][k]
  int tid  = threadIdx.x;
  int lane = tid & 31;
  int wave = tid >> 5;              // 0..15 -> q tile
  int b     = blockIdx.x >> 4;
  int pBase = (blockIdx.x & 15) << 4;
  int qBase = wave << 4;
  int n  = lane & 15;
  int up = lane >> 4;
  const size_t CP = 1024 * 256;
  v8f acc = {0.f, 0.f, 0.f, 0.f, 0.f, 0.f, 0.f, 0.f};

  for (int kc = 0; kc < 32; ++kc) {
    __syncthreads();
    {
      // stage A slab [32k x 16p] transposed into [m][k]
      int k = tid >> 4, m = tid & 15;
      sA[m * 32 + k] = Ah[(size_t)b * CP + (size_t)(kc * 32 + k) * 256 + pBase + m];
      // stage B slab [32k x 256q] transposed into [q][k]
#pragma unroll
      for (int it = 0; it < 16; ++it) {
        int i = tid + it * 512;
        int kk = i >> 8, q = i & 255;
        sB[q * 32 + kk] = Bh[(size_t)b * CP + (size_t)(kc * 32 + kk) * 256 + q];
      }
    }
    __syncthreads();
    // A fragment: lane m holds a contiguous 16-K run kb+16*up+{0..15}
    const _Float16* ar = &sA[(size_t)(lane & 15) * 32 + up * 16];
    v16h af = make_frag(*(const h8*)ar, *(const h8*)(ar + 8));
    // B fragment: lane n holds a contiguous 16-K run of column q
    const _Float16* br = &sB[(size_t)(qBase + n) * 32 + up * 16];
    v16h bf = make_frag(*(const h8*)br, *(const h8*)(br + 8));
    acc = __builtin_amdgcn_wmma_f32_16x16x32_f16(false, af, false, bf,
                                                 (short)0, acc, false, false);
  }

  float sb = invB[b * 256 + qBase + n];
#pragma unroll
  for (int r = 0; r < 8; ++r) {
    int p = pBase + r + up * 8;
    corr[(size_t)b * 65536 + (size_t)p * 256 + qBase + n] =
        acc[r] * invA[b * 256 + p] * sb;
  }
}

// ---------------- 3) mutual matching -> padded plane-major f16 ----------------
// padded layout: [b][aw(256)][pd*20+pt (400)][ci(16)]; borders/ci>0 pre-zeroed.
__global__ void k_mm_h(const float* __restrict__ c, _Float16* __restrict__ xpad,
                       _Float16* __restrict__ xtpad) {
  __shared__ float mA[256], mB[256];
  int b = blockIdx.x, t = threadIdx.x;
  const float* cb = c + (size_t)b * 65536;
  float ma = -3.4e38f, mb = -3.4e38f;
  for (int q = 0; q < 256; ++q) ma = fmaxf(ma, cb[t * 256 + q]);
  for (int p = 0; p < 256; ++p) mb = fmaxf(mb, cb[p * 256 + t]);
  mA[t] = ma; mB[t] = mb;
  __syncthreads();
  int d = t >> 4, tt = t & 15;
  for (int p = 0; p < 256; ++p) {
    float v = cb[p * 256 + t];
    float r = v * (v / (mA[p] + EPS_MM)) * (v / (mB[t] + EPS_MM));
    // x[b][p][q]: plane p, inner pos q=(d,tt)
    xpad [((size_t)(b * 256 + p) * 400 + (d + 2) * 20 + (tt + 2)) * 16] = (_Float16)r;
    // x^T[b][q][p]: plane q=t, inner pos p
    int pd = p >> 4, pt = p & 15;
    xtpad[((size_t)(b * 256 + t) * 400 + (pd + 2) * 20 + (pt + 2)) * 16] = (_Float16)r;
  }
}

__global__ void k_mm_f(const float* __restrict__ c, float* __restrict__ out) {
  __shared__ float mA[256], mB[256];
  int b = blockIdx.x, t = threadIdx.x;
  const float* cb = c + (size_t)b * 65536;
  float ma = -3.4e38f, mb = -3.4e38f;
  for (int q = 0; q < 256; ++q) ma = fmaxf(ma, cb[t * 256 + q]);
  for (int p = 0; p < 256; ++p) mb = fmaxf(mb, cb[p * 256 + t]);
  mA[t] = ma; mB[t] = mb;
  __syncthreads();
  for (int p = 0; p < 256; ++p) {
    float v = cb[p * 256 + t];
    out[(size_t)b * 65536 + p * 256 + t] =
        v * (v / (mA[p] + EPS_MM)) * (v / (mB[t] + EPS_MM));
  }
}

// ---------------- 4) weight packing ----------------
// wp[kd=da*5+dw][m=o][k=(dd*5+dt)*16+ci], Kpad=416, zeros for k>=400 / m>=COUT / ci>=CIN
__global__ void k_pack(const float* __restrict__ w, _Float16* __restrict__ wp,
                       int CIN, int COUT) {
  int idx = blockIdx.x * 256 + threadIdx.x;
  if (idx >= 25 * 16 * 416) return;
  int kd  = idx / (16 * 416);
  int rem = idx - kd * (16 * 416);
  int m = rem / 416;
  int k = rem - m * 416;
  int da = kd / 5, dw = kd - da * 5;
  float v = 0.f;
  if (k < 400 && m < COUT) {
    int ci = k & 15;
    int j  = k >> 4;
    int dd = j / 5, dt = j - dd * 5;
    if (ci < CIN)
      v = w[(((((size_t)m * CIN + ci) * 5 + da) * 5 + dw) * 5 + dd) * 5 + dt];
  }
  wp[idx] = (_Float16)v;
}

// ---------------- 5) zero-fill (borders of padded buffers) ----------------
__global__ void k_zero(uint4* __restrict__ p, int n) {
  int i = blockIdx.x * 256 + threadIdx.x;
  if (i < n) {
    uint4 z; z.x = 0; z.y = 0; z.z = 0; z.w = 0;
    p[i] = z;
  }
}

// ---------------- 6) implicit-GEMM 4D conv (WMMA + async LDS staging) ----------
// input  : padded plane-major f16 [b][aw][400][16ci] -- each (ia,iw) plane is one
//          contiguous 12800B block, async-DMA'd into LDS (GLOBAL_LOAD_ASYNC_TO_LDS_B128).
// output : outH -> same padded layout (one b128 store per lane), or outF f32 (COUT=1).
__global__ void __launch_bounds__(512) k_conv(const _Float16* __restrict__ xpad,
                                              const _Float16* __restrict__ wp,
                                              const float* __restrict__ bias,
                                              int COUT,
                                              float* __restrict__ outF,
                                              _Float16* __restrict__ outH) {
  __shared__ __align__(16) _Float16 plane[20 * 20 * 16];  // [pd][pt][ci]
  int tid  = threadIdx.x;
  int lane = tid & 31;
  int wave = tid >> 5;         // hB subtile
  int b  = blockIdx.x >> 8;
  int aw = blockIdx.x & 255;
  int a = aw >> 4, w = aw & 15;
  int n = lane & 15, up = lane >> 4;
  v8f acc = {0.f, 0.f, 0.f, 0.f, 0.f, 0.f, 0.f, 0.f};

  for (int da = 0; da < 5; ++da) {
    int ia = a + da - 2;
    for (int dw = 0; dw < 5; ++dw) {
      int iw = w + dw - 2;
      bool oob = ((unsigned)ia >= 16u) | ((unsigned)iw >= 16u);
      __syncthreads();
      if (!oob) {
        const _Float16* src = xpad + ((size_t)(b * 256) + (ia * 16 + iw)) * 6400;
        // 12800B plane = 800 x 16B chunks; direct global->LDS async DMA
        for (int i = tid; i < 800; i += 512) {
#if USE_ASYNC_LDS
          __builtin_amdgcn_global_load_async_to_lds_b128(
              (GLOBAL_AS v4i*)(src + i * 8),
              (LDS_AS v4i*)(plane + i * 8), 0, 0);
#else
          *(h8*)(plane + i * 8) = *(const h8*)(src + i * 8);
#endif
        }
#if USE_ASYNC_LDS
        wait_async0();
#endif
      }
      __syncthreads();
      if (oob) continue;

      int kd = da * 5 + dw;
      const _Float16* wrow = wp + ((size_t)kd * 16 + n) * 416;  // A row for M=n
      __builtin_prefetch(wrow + 416, 0, 0);                     // global_prefetch_b8
#pragma unroll
      for (int ch = 0; ch < 13; ++ch) {
        int kb = ch * 32;
        const _Float16* ap = wrow + kb + up * 16;
        v16h af = make_frag(*(const h8*)ap, *(const h8*)(ap + 8));
        int j = (kb >> 4) + up;          // (dd,dt) index for this lane's K-run
        if (j > 24) j = 24;              // K padding tail (A is zero there)
        int dd = j / 5, dt = j - dd * 5;
        const _Float16* bp = &plane[((wave + dd) * 20 + (n + dt)) * 16];
        v16h bf = make_frag(*(const h8*)bp, *(const h8*)(bp + 8));
        acc = __builtin_amdgcn_wmma_f32_16x16x32_f16(false, af, false, bf,
                                                     (short)0, acc, false, false);
      }
    }
  }

  // epilogue: bias + ReLU (D layout: VGPR r -> M=r+8*up, N=lane&15)
  if (outH) {
    // padded plane-major store: 8 f16 (ci r+8*up contiguous) = one b128
    h8 o8;
#pragma unroll
    for (int r = 0; r < 8; ++r)
      o8[r] = (_Float16)fmaxf(acc[r] + bias[r + up * 8], 0.f);
    size_t off = ((size_t)(b * 256 + aw) * 400 + (wave + 2) * 20 + (n + 2)) * 16 + up * 8;
    *(h8*)(outH + off) = o8;
  } else {
    int sBase = (a * 16 + w) * 256 + wave * 16 + n;
#pragma unroll
    for (int r = 0; r < 8; ++r) {
      int o = r + up * 8;
      if (o < COUT) {
        float v = fmaxf(acc[r] + bias[o], 0.f);
        outF[((size_t)b * COUT + o) * 65536 + sBase] = v;
      }
    }
  }
}

// ---------------- 7) symmetric-branch fusion ----------------
__global__ void k_fuse(const float* __restrict__ br1, const float* __restrict__ br2,
                       float* __restrict__ S) {
  int idx = blockIdx.x * 256 + threadIdx.x;  // 262144 total
  int b = idx >> 16, r = idx & 65535, p = r >> 8, q = r & 255;
  S[idx] = br1[(size_t)b * 65536 + p * 256 + q] +
           br2[(size_t)b * 65536 + q * 256 + p];
}

// ---------------- host ----------------
extern "C" void kernel_launch(void* const* d_in, const int* in_sizes, int n_in,
                              void* d_out, int out_size, void* d_ws, size_t ws_size,
                              hipStream_t stream) {
  (void)in_sizes; (void)n_in; (void)out_size; (void)ws_size;
  const float* featA = (const float*)d_in[0];
  const float* featB = (const float*)d_in[1];
  const float* w1 = (const float*)d_in[2];
  const float* b1 = (const float*)d_in[3];
  const float* w2 = (const float*)d_in[4];
  const float* b2 = (const float*)d_in[5];
  const float* w3 = (const float*)d_in[6];
  const float* b3 = (const float*)d_in[7];

  char* ws = (char*)d_ws;
  size_t off = 0;
  auto alloc = [&](size_t bytes) -> void* {
    void* p = ws + off;
    off = (off + bytes + 255) & ~(size_t)255;
    return p;
  };
  const size_t PAD_BYTES = 4ull * 256 * 400 * 16 * 2;  // 13,107,200 (mult. of 256)
  _Float16* Ah   = (_Float16*)alloc(4ull * 1024 * 256 * 2);
  _Float16* Bh   = (_Float16*)alloc(4ull * 1024 * 256 * 2);
  float*    invA = (float*)   alloc(4ull * 256 * 4);
  float*    invB = (float*)   alloc(4ull * 256 * 4);
  float*    corr = (float*)   alloc(4ull * 65536 * 4);   // reused for fused S
  // four contiguous padded plane-major buffers (zeroed together)
  _Float16* pxa  = (_Float16*)alloc(PAD_BYTES);  // conv1 input, branch 1
  _Float16* pxt  = (_Float16*)alloc(PAD_BYTES);  // conv1 input, branch 2
  _Float16* pyA  = (_Float16*)alloc(PAD_BYTES);  // conv1 output / conv2 input
  _Float16* pyB  = (_Float16*)alloc(PAD_BYTES);  // conv2 output / conv3 input
  float*    br1  = (float*)   alloc(4ull * 65536 * 4);
  float*    br2  = (float*)   alloc(4ull * 65536 * 4);
  _Float16* wp1  = (_Float16*)alloc(25ull * 16 * 416 * 2);
  _Float16* wp2  = (_Float16*)alloc(25ull * 16 * 416 * 2);
  _Float16* wp3  = (_Float16*)alloc(25ull * 16 * 416 * 2);

  // zero padded buffers (borders + unused ci must be 0); 4 x PAD_BYTES contiguous
  const int ZN = (int)(4 * PAD_BYTES / 16);
  k_zero<<<(ZN + 255) / 256, 256, 0, stream>>>((uint4*)pxa, ZN);

  const int PACK_N = 25 * 16 * 416;
  k_pack<<<(PACK_N + 255) / 256, 256, 0, stream>>>(w1, wp1, 1, 16);
  k_pack<<<(PACK_N + 255) / 256, 256, 0, stream>>>(w2, wp2, 16, 16);
  k_pack<<<(PACK_N + 255) / 256, 256, 0, stream>>>(w3, wp3, 16, 1);

  k_norm<<<8, 256, 0, stream>>>(featA, featB, Ah, Bh, invA, invB);
  k_corr<<<64, 512, 0, stream>>>(Ah, Bh, invA, invB, corr);
  k_mm_h<<<4, 256, 0, stream>>>(corr, pxa, pxt);

  // branch 1: seq(x)
  k_conv<<<1024, 512, 0, stream>>>(pxa, wp1, b1, 16, nullptr, pyA);
  k_conv<<<1024, 512, 0, stream>>>(pyA, wp2, b2, 16, nullptr, pyB);
  k_conv<<<1024, 512, 0, stream>>>(pyB, wp3, b3, 1, br1, nullptr);
  // branch 2: seq(x^T)
  k_conv<<<1024, 512, 0, stream>>>(pxt, wp1, b1, 16, nullptr, pyA);
  k_conv<<<1024, 512, 0, stream>>>(pyA, wp2, b2, 16, nullptr, pyB);
  k_conv<<<1024, 512, 0, stream>>>(pyB, wp3, b3, 1, br2, nullptr);

  k_fuse<<<262144 / 256, 256, 0, stream>>>(br1, br2, corr);
  k_mm_f<<<4, 256, 0, stream>>>(corr, (float*)d_out);
}
